// MultiHeadAttention_12214886989963
// MI455X (gfx1250) — matrix operand order repaired and checked by reading, not compiled, not run on hardware
//
#include <hip/hip_runtime.h>

#define S_LEN  2048
#define NHEAD  16
#define HDIM   64
#define DMODEL 1024
#define BATCH  2

typedef __attribute__((ext_vector_type(16))) _Float16     v16h;
typedef __attribute__((ext_vector_type(8)))  _Float16     v8h;
typedef __attribute__((ext_vector_type(4)))  _Float16     v4h;
typedef __attribute__((ext_vector_type(8)))  float        v8f;
typedef __attribute__((ext_vector_type(8)))  unsigned int v8u;

static __device__ __forceinline__ v16h cat8(v8h lo, v8h hi) {
  return __builtin_shufflevector(lo, hi, 0,1,2,3,4,5,6,7,8,9,10,11,12,13,14,15);
}

// D = A(16x32 f16) x B(32x16 f16) + C(16x16 f32)
static __device__ __forceinline__ v8f wmma16x16x32(v16h a, v16h b, v8f c) {
  return __builtin_amdgcn_wmma_f32_16x16x32_f16(false, a, false, b, (short)0, c, false, false);
}

static __device__ __forceinline__ unsigned pack2h(float a, float b) {
  union { _Float16 h[2]; unsigned u; } t;
  t.h[0] = (_Float16)a; t.h[1] = (_Float16)b;
  return t.u;
}

// ---------------------------------------------------------------- convert
__global__ void cvt_f32_f16_k(const float* __restrict__ in,
                              _Float16* __restrict__ out, int n) {
  int i = (blockIdx.x * blockDim.x + threadIdx.x) * 4;
  if (i + 3 < n) {
    const float4 f = *(const float4*)(in + i);
    v4h h; h.x = (_Float16)f.x; h.y = (_Float16)f.y;
           h.z = (_Float16)f.z; h.w = (_Float16)f.w;
    *(v4h*)(out + i) = h;
  }
}

// ---------------------------------------------------------------- GEMM  Y = X * W^T + bias
// One wave computes a 32x64 output tile (2 M-frags x 4 N-frags = 8 WMMAs per
// K-step of 32; 12 b128 loads per 8 WMMAs).
// mode 0: store f16 to [B,H,S,HD]   (Q, K)
// mode 1: store f16 to [B,H,HD,S]   (V transposed)
// mode 2: store f32 to [rows, cols] (final output)
__global__ __launch_bounds__(128) void gemm_xwT_wmma(
    const _Float16* __restrict__ A, const _Float16* __restrict__ W,
    const float* __restrict__ bias, void* __restrict__ outp,
    int rows, int cols, int kdim, int mode)
{
  const int lane = threadIdx.x & 31;
  const int lm   = lane & 15;     // M (A) / N (B) index within 16
  const int hi   = lane >> 4;     // lane-group
  const int wave = blockIdx.x * (blockDim.x >> 5) + (threadIdx.x >> 5);
  const int ntc  = cols >> 6;
  const int s0   = (wave / ntc) << 5;
  const int e0   = (wave % ntc) << 6;

  v8f acc[2][4] = {};

  for (int kk = 0; kk < kdim; kk += 32) {
    v16h a[2], bb[4];
#pragma unroll
    for (int mt = 0; mt < 2; ++mt) {
      // A fragment: lane holds row (s0+mt*16+lm), K = hi*16..+15 contiguous
      const _Float16* pa = A + (size_t)(s0 + mt*16 + lm) * kdim + kk + hi*16;
      a[mt] = cat8(*(const v8h*)pa, *(const v8h*)(pa + 8));
    }
#pragma unroll
    for (int nt = 0; nt < 4; ++nt) {
      // B fragment: B[k][n] = W[e0+nt*16+n][kk+k]; lane holds 16 contiguous K
      const _Float16* pw = W + (size_t)(e0 + nt*16 + lm) * kdim + kk + hi*16;
      bb[nt] = *(const v16h*)pw;
    }
#pragma unroll
    for (int mt = 0; mt < 2; ++mt)
#pragma unroll
      for (int nt = 0; nt < 4; ++nt)
        acc[mt][nt] = wmma16x16x32(a[mt], bb[nt], acc[mt][nt]);
  }

  // D fragment: lane has N = lm, M = hi*8 + v
#pragma unroll
  for (int nt = 0; nt < 4; ++nt) {
    const int e = e0 + nt*16 + lm;
    const float bval = bias[e];
    const int h = e >> 6, d = e & 63;
#pragma unroll
    for (int mt = 0; mt < 2; ++mt) {
#pragma unroll
      for (int v = 0; v < 8; ++v) {
        const int r = s0 + mt*16 + hi*8 + v;
        const float val = acc[mt][nt][v] + bval;
        if (mode == 2) {
          ((float*)outp)[(size_t)r * cols + e] = val;
        } else {
          const int b = r >> 11, s = r & (S_LEN - 1);
          _Float16* O = (_Float16*)outp;
          if (mode == 0)
            O[(((size_t)(b*NHEAD + h)) * S_LEN + s) * HDIM + d] = (_Float16)val;
          else
            O[(((size_t)(b*NHEAD + h)) * HDIM + d) * S_LEN + s] = (_Float16)val;
        }
      }
    }
  }
}

// ---------------------------------------------------------------- flash attention
// One wave per (b,h,q-tile of 32 rows). Computes scores^T = K*Q^T so softmax
// over kv is per-lane + one xor-16 shuffle, and out^T = V^T * P^T.
// Two q-halves share every K/V fragment load: 16 b128 loads per 16 WMMAs/chunk.
__global__ __launch_bounds__(128) void flash_attn_wmma(
    const _Float16* __restrict__ Q, const _Float16* __restrict__ K,
    const _Float16* __restrict__ VT, _Float16* __restrict__ O)
{
  const int lane = threadIdx.x & 31;
  const int lm   = lane & 15;
  const int hi   = lane >> 4;
  const int wave = blockIdx.x * (blockDim.x >> 5) + (threadIdx.x >> 5);
  const int qt   = wave & 63;
  const int bh   = wave >> 6;          // b*16 + h
  const int q0   = qt << 5;

  const _Float16* Qb = Q  + (size_t)bh * S_LEN * HDIM;
  const _Float16* Kb = K  + (size_t)bh * S_LEN * HDIM;
  const _Float16* Vb = VT + (size_t)bh * HDIM * S_LEN;

  // B fragments of Q^T (d x q) for both 16-row q-halves
  v16h bq[2][2];
#pragma unroll
  for (int qh = 0; qh < 2; ++qh)
#pragma unroll
    for (int dh = 0; dh < 2; ++dh)
      bq[qh][dh] = cat8(*(const v8h*)(Qb + (size_t)(q0 + qh*16 + lm) * HDIM + dh*32 + hi*8), *(const v8h*)(Qb + (size_t)(q0 + qh*16 + lm) * HDIM + dh*32 + hi*8 + 16));

  v8f o[2][4] = {};                 // out^T accumulators per q-half
  float mi[2] = {-3.0e38f, -3.0e38f};
  float li[2] = {0.f, 0.f};
  const int nc = (q0 + 32) >> 5;    // exact: last chunk is the diagonal block
  const float LOG2E = 1.4426950408889634f;

  for (int c = 0; c < nc; ++c) {
    const int kc = c << 5;
    const bool lastc = (c == nc - 1);

    // K fragments, shared by both q-halves
    v16h aK[2][2];
#pragma unroll
    for (int st = 0; st < 2; ++st) {
      const _Float16* pk = Kb + (size_t)(kc + st*16 + lm) * HDIM + hi*8;
      aK[st][0] = cat8(*(const v8h*)pk,        *(const v8h*)(pk + 16));
      aK[st][1] = cat8(*(const v8h*)(pk + 32), *(const v8h*)(pk + 48));
    }

    // scores^T (kv x q): 8 WMMAs
    v8f sf[2][2];
#pragma unroll
    for (int qh = 0; qh < 2; ++qh)
#pragma unroll
      for (int st = 0; st < 2; ++st) {
        v8f z = {};
        z           = wmma16x16x32(aK[st][0], bq[qh][0], z);
        sf[qh][st]  = wmma16x16x32(aK[st][1], bq[qh][1], z);
      }

    // scale (+ causal mask on the diagonal chunk only), online softmax
    v16h bp[2];
#pragma unroll
    for (int qh = 0; qh < 2; ++qh) {
      const int qg = q0 + qh*16 + lm;
      float mloc = -3.0e38f;
#pragma unroll
      for (int st = 0; st < 2; ++st)
#pragma unroll
        for (int v = 0; v < 8; ++v) {
          float xv = sf[qh][st][v] * 0.125f;
          if (lastc) {
            const int kv = kc + st*16 + hi*8 + v;
            xv = (kv <= qg) ? xv : -1.0e9f;
          }
          sf[qh][st][v] = xv;
          mloc = fmaxf(mloc, xv);
        }
      const float mx   = fmaxf(mloc, __shfl_xor(mloc, 16, 32));
      const float mnew = fmaxf(mi[qh], mx);
      const float rsc  = exp2f((mi[qh] - mnew) * LOG2E);
      float lloc = 0.f;
#pragma unroll
      for (int st = 0; st < 2; ++st)
#pragma unroll
        for (int v = 0; v < 8; ++v) {
          const float p = exp2f((sf[qh][st][v] - mnew) * LOG2E);
          sf[qh][st][v] = p;
          lloc += p;
        }
      li[qh] = li[qh] * rsc + lloc + __shfl_xor(lloc, 16, 32);
#pragma unroll
      for (int dt = 0; dt < 4; ++dt) o[qh][dt] *= rsc;
      mi[qh] = mnew;

      // Convert P^T from D-layout to B-layout via 8 xor-16 shuffles
      unsigned p0[4], p1[4];
#pragma unroll
      for (int j = 0; j < 4; ++j) {
        p0[j] = pack2h(sf[qh][0][2*j], sf[qh][0][2*j + 1]);
        p1[j] = pack2h(sf[qh][1][2*j], sf[qh][1][2*j + 1]);
      }
      v8u bw;
#pragma unroll
      for (int j = 0; j < 4; ++j) {
        const unsigned x0 = __shfl_xor(p0[j], 16, 32);
        const unsigned x1 = __shfl_xor(p1[j], 16, 32);
        bw[j]     = hi ? x1    : p0[j];
        bw[4 + j] = hi ? p1[j] : x0;
      }
      bp[qh] = __builtin_bit_cast(v16h, bw);
    }

    // out^T += V^T(16d x 32kv) * P^T(32kv x 16q); V fragments shared: 8 WMMAs
#pragma unroll
    for (int dt = 0; dt < 4; ++dt) {
      const _Float16* pv = Vb + (size_t)(dt*16 + lm) * S_LEN + kc + hi*16;
      const v16h av = cat8(*(const v8h*)pv, *(const v8h*)(pv + 8));
#pragma unroll
      for (int qh = 0; qh < 2; ++qh)
        o[qh][dt] = wmma16x16x32(av, bp[qh], o[qh][dt]);
    }

    if (c + 1 < nc) {
      __builtin_prefetch(Kb + (size_t)(kc + 32 + lm) * HDIM, 0, 1);
      __builtin_prefetch(Vb + (size_t)lm * S_LEN + kc + 32, 0, 1);
    }
  }

  // normalize and store concat-head layout [B,S,D] as f16
  const int b = bh >> 4, h = bh & 15;
#pragma unroll
  for (int qh = 0; qh < 2; ++qh) {
    const float inv = 1.0f / li[qh];
    _Float16* Orow = O + (((size_t)b * S_LEN) + q0 + qh*16 + lm) * DMODEL + h * HDIM;
#pragma unroll
    for (int dt = 0; dt < 4; ++dt)
#pragma unroll
      for (int v = 0; v < 8; ++v)
        Orow[dt*16 + hi*8 + v] = (_Float16)(o[qh][dt][v] * inv);
  }
}

// ---------------------------------------------------------------- launch
extern "C" void kernel_launch(void* const* d_in, const int* in_sizes, int n_in,
                              void* d_out, int out_size, void* d_ws, size_t ws_size,
                              hipStream_t stream)
{
  (void)in_sizes; (void)n_in; (void)out_size; (void)ws_size;
  const float* x  = (const float*)d_in[0];
  // d_in[1] is the causal mask; setup guarantees tril, implemented analytically.
  const float* Wq = (const float*)d_in[2];
  const float* bq = (const float*)d_in[3];
  const float* Wk = (const float*)d_in[4];
  const float* bk = (const float*)d_in[5];
  const float* Wv = (const float*)d_in[6];
  const float* bv = (const float*)d_in[7];
  const float* Wo = (const float*)d_in[8];
  const float* bo = (const float*)d_in[9];

  char* ws = (char*)d_ws;
  const size_t MB = 1024u * 1024u;
  _Float16* xh  = (_Float16*)(ws);             //  8 MB
  _Float16* Wqh = (_Float16*)(ws +  8*MB);     //  2 MB
  _Float16* Wkh = (_Float16*)(ws + 10*MB);     //  2 MB
  _Float16* Wvh = (_Float16*)(ws + 12*MB);     //  2 MB
  _Float16* Woh = (_Float16*)(ws + 14*MB);     //  2 MB
  _Float16* Qh  = (_Float16*)(ws + 16*MB);     //  8 MB [B,H,S,HD]
  _Float16* Kh  = (_Float16*)(ws + 24*MB);     //  8 MB [B,H,S,HD]
  _Float16* VTh = (_Float16*)(ws + 32*MB);     //  8 MB [B,H,HD,S]
  _Float16* Oh  = (_Float16*)(ws + 40*MB);     //  8 MB [B,S,D]

  const int nx = BATCH * S_LEN * DMODEL;       // 4,194,304
  const int nw = DMODEL * DMODEL;              // 1,048,576
  cvt_f32_f16_k<<<nx / 1024, 256, 0, stream>>>(x,  xh,  nx);
  cvt_f32_f16_k<<<nw / 1024, 256, 0, stream>>>(Wq, Wqh, nw);
  cvt_f32_f16_k<<<nw / 1024, 256, 0, stream>>>(Wk, Wkh, nw);
  cvt_f32_f16_k<<<nw / 1024, 256, 0, stream>>>(Wv, Wvh, nw);
  cvt_f32_f16_k<<<nw / 1024, 256, 0, stream>>>(Wo, Woh, nw);

  const int rows = BATCH * S_LEN;                            // 4096
  const int gemmBlocks = (rows / 32) * (DMODEL / 64) / 4;    // 512 blocks of 4 waves
  gemm_xwT_wmma<<<gemmBlocks, 128, 0, stream>>>(xh, Wqh, bq, (void*)Qh,  rows, DMODEL, DMODEL, 0);
  gemm_xwT_wmma<<<gemmBlocks, 128, 0, stream>>>(xh, Wkh, bk, (void*)Kh,  rows, DMODEL, DMODEL, 0);
  gemm_xwT_wmma<<<gemmBlocks, 128, 0, stream>>>(xh, Wvh, bv, (void*)VTh, rows, DMODEL, DMODEL, 1);

  const int attnBlocks = (BATCH * NHEAD * (S_LEN / 32)) / 4; // 512
  flash_attn_wmma<<<attnBlocks, 128, 0, stream>>>(Qh, Kh, VTh, Oh);

  gemm_xwT_wmma<<<gemmBlocks, 128, 0, stream>>>(Oh, Woh, bo, d_out, rows, DMODEL, DMODEL, 2);
}
